// TemplatePairStackBlock_81149112091144
// MI455X (gfx1250) — compile-verified, matrix operand order checked by reading, not run on hardware
//
#include <hip/hip_runtime.h>
#include <hip/hip_bf16.h>

// ---------------------------------------------------------------------------
// CDNA5 / gfx1250 WMMA helpers (wave32, V_WMMA_F32_16X16X32_F16)
// ---------------------------------------------------------------------------
typedef __attribute__((ext_vector_type(16))) _Float16 v16h;
typedef __attribute__((ext_vector_type(8)))  float    v8f;
typedef __attribute__((ext_vector_type(4)))  unsigned int u32x4;
typedef __attribute__((ext_vector_type(8)))  unsigned int u32x8;

#define TT 2
#define NN 256
#define CC 64
#define HH 4
#define CHD 16

__device__ inline v8f wmma16(v16h a, v16h b, v8f c) {
  // D = A(16x32 f16) * B(32x16 f16) + C(16x16 f32)
  return __builtin_amdgcn_wmma_f32_16x16x32_f16(false, a, false, b, (short)0, c,
                                                false, false);
}

// Generic 16-bit fragment loader.
// Per ISA 7.12.2 (16-bit A 16x32): lane L -> fixed index (M or N) = L%16,
// hi = L/16; VGPR d holds K pair:
//   K(d, hi) = (d>=4 ? 16 : 0) + (d%4)*2 + hi*8   (+0/+1 for the two halves)
// A row-major [16][lda]          : frag_ld(base, lda, 1)
// A^T stored  [k][ld], M in cols : frag_ld(base, 1,  ld)
// B row-major [32][ldb], N cols  : frag_ld(base, 1,  ldb)
// B^T stored  [n][ld], K in cols : frag_ld(base, ld, 1)
__device__ inline v16h frag_ld(const _Float16* p0, int sr, int sc) {
  int lane = threadIdx.x & 31;
  int fix = lane & 15, hi = lane >> 4;
  const _Float16* p = p0 + (size_t)fix * sr;
  v16h r;
#pragma unroll
  for (int d = 0; d < 8; ++d) {
    int kb = ((d & 4) << 2) + ((d & 3) << 1) + (hi << 3);
    r[2 * d]     = p[(size_t)kb * sc];
    r[2 * d + 1] = p[(size_t)(kb + 1) * sc];
  }
  return r;
}

__device__ inline v16h frag_ld_f32(const float* p0, int sr, int sc) {
  int lane = threadIdx.x & 31;
  int fix = lane & 15, hi = lane >> 4;
  const float* p = p0 + (size_t)fix * sr;
  v16h r;
#pragma unroll
  for (int d = 0; d < 8; ++d) {
    int kb = ((d & 4) << 2) + ((d & 3) << 1) + (hi << 3);
    r[2 * d]     = (_Float16)p[(size_t)kb * sc];
    r[2 * d + 1] = (_Float16)p[(size_t)(kb + 1) * sc];
  }
  return r;
}

__device__ inline float sigmoidf(float x) { return 1.0f / (1.0f + __expf(-x)); }

__device__ inline void stage_w(const float* W, _Float16* wsm, int n, int tid) {
  for (int idx = tid; idx < n; idx += 256) wsm[idx] = (_Float16)W[idx];
}

// OUT[256][64] = op( XLN[256][64](f16) @ W[64][64](f16 LDS) [+ bvec] )
// mode: 0 = plain(+bvec), 1 = sigmoid(x+bvec), 2 = x*0.25
__device__ inline void projG(const _Float16* xln, const _Float16* wsm,
                             _Float16* dst, int mode, const float* bvec) {
  int wid = threadIdx.x >> 5;
  int lane = threadIdx.x & 31, n = lane & 15, hi = lane >> 4;
  for (int tl = wid; tl < 64; tl += 8) {
    int jt = tl >> 2, nt = tl & 3;
    v8f acc = {};
#pragma unroll
    for (int kk = 0; kk < 64; kk += 32)
      acc = wmma16(frag_ld(xln + jt * 16 * 64 + kk, 64, 1),
                   frag_ld(wsm + kk * 64 + nt * 16, 1, 64), acc);
    int col = nt * 16 + n;
#pragma unroll
    for (int r = 0; r < 8; ++r) {
      int j = jt * 16 + r + hi * 8;
      float v = acc[r] + (bvec ? bvec[col] : 0.0f);
      if (mode == 2) v = acc[r] * 0.25f;            // 1/sqrt(CH=16)
      else if (mode == 1) v = sigmoidf(v);
      dst[(size_t)j * 64 + col] = (_Float16)v;
    }
  }
}

__device__ inline void layernorm_row(const float* row, const float* ls,
                                     const float* lb, _Float16* out) {
  float mean = 0.f;
  for (int c = 0; c < 64; ++c) mean += row[c];
  mean *= (1.0f / 64.0f);
  float var = 0.f;
  for (int c = 0; c < 64; ++c) { float d = row[c] - mean; var += d * d; }
  var *= (1.0f / 64.0f);
  float rs = rsqrtf(var + 1e-5f);
  for (int c = 0; c < 64; ++c)
    out[c] = (_Float16)((row[c] - mean) * rs * ls[c] + lb[c]);
}

// ---------------------------------------------------------------------------
// Tensor Data Mover (TDM): 2-D tile DMA global -> LDS (ISA 08_async_tensor).
// Descriptor group0 (4 SGPRs) + group1 (8 SGPRs); issued once per wave;
// completion fenced with s_wait_tensorcnt.
// ---------------------------------------------------------------------------
__device__ inline u32x4 tdm_g0(unsigned lds_addr, unsigned long long gaddr) {
  u32x4 g;
  g[0] = 1u;                                          // count=1, user mode
  g[1] = lds_addr;                                    // lds_addr [63:32]
  g[2] = (unsigned)(gaddr & 0xffffffffull);           // global_addr[31:0]
  g[3] = (unsigned)((gaddr >> 32) & 0x01fffffful)     // global_addr[56:32]
       | (2u << 30);                                  // type=2 ("image")
  return g;
}

// w0: data_size=1B; optional LDS padding (pad after every 8 DWORDs data).
// td0/td1 = tensor dims (bytes / rows); tile0/tile1 = tile dims;
// stride0 = row pitch in bytes.
__device__ inline u32x8 tdm_g1(unsigned w0, unsigned td0, unsigned td1,
                               unsigned tile0, unsigned tile1,
                               unsigned long long stride0) {
  u32x8 g;
  g[0] = w0;
  g[1] = (td0 & 0xffffu) << 16;                       // atomic_addr=0 | td0 lo
  g[2] = (td0 >> 16) | ((td1 & 0xffffu) << 16);
  g[3] = (td1 >> 16) | (tile0 << 16);
  g[4] = tile1 & 0xffffu;                             // tile_dim2 = 0
  g[5] = (unsigned)(stride0 & 0xffffffffull);
  g[6] = (unsigned)((stride0 >> 32) & 0xffffull);     // dim1_stride = 0
  g[7] = 0;
  return g;
}

__device__ inline void tdm_load(u32x4 g0, u32x8 g1) {
  asm volatile("tensor_load_to_lds %0, %1" :: "s"(g0), "s"(g1) : "memory");
}

// pad_enable | pad_interval=8dw (code 2) | pad_amount=8dw (code 7)
#define TDM_PAD_W0 ((1u << 20) | (2u << 22) | (7u << 25))

// ---------------------------------------------------------------------------
// Triangle attention: phase A (LN + projections)
// ---------------------------------------------------------------------------
struct AttnPrepP {
  const float* s;
  const float* ln_s; const float* ln_b;
  const float* wq; const float* wk; const float* wv; const float* wb;
  const float* wg; const float* bg;
  _Float16* qf; _Float16* kf; _Float16* vf; _Float16* gf;
  float* bias;   // [T][H][N][N]
  int tr;
};

extern "C" __global__ __launch_bounds__(256) void k_attn_prep(AttnPrepP p) {
  extern __shared__ char smem[];
  _Float16* xln = (_Float16*)smem;     // [256][64]
  _Float16* wsm = xln + NN * CC;       // [64][64]
  int t = blockIdx.x >> 8, i = blockIdx.x & 255;
  int tid = threadIdx.x;
  {
    int j = tid;
    const float* row = p.s +
        (((size_t)t * NN + (p.tr ? j : i)) * NN + (p.tr ? i : j)) * CC;
    layernorm_row(row, p.ln_s, p.ln_b, xln + (size_t)j * 64);
  }
  __syncthreads();
  size_t rowbase = (((size_t)t * NN + i) * NN) * CC;

  stage_w(p.wq, wsm, 64 * 64, tid); __syncthreads();
  projG(xln, wsm, p.qf + rowbase, 2, nullptr); __syncthreads();
  stage_w(p.wk, wsm, 64 * 64, tid); __syncthreads();
  projG(xln, wsm, p.kf + rowbase, 0, nullptr); __syncthreads();
  stage_w(p.wv, wsm, 64 * 64, tid); __syncthreads();
  projG(xln, wsm, p.vf + rowbase, 0, nullptr); __syncthreads();
  stage_w(p.wg, wsm, 64 * 64, tid); __syncthreads();
  projG(xln, wsm, p.gf + rowbase, 1, p.bg); __syncthreads();

  // tri_bias: wb [64][4] zero-padded to [64][16]
  for (int idx = tid; idx < 64 * 16; idx += 256) {
    int c = idx >> 4, hcol = idx & 15;
    wsm[idx] = (hcol < 4) ? (_Float16)p.wb[c * 4 + hcol] : (_Float16)0.f;
  }
  __syncthreads();
  int wid = tid >> 5, lane = tid & 31, n = lane & 15, hi = lane >> 4;
  for (int jt = wid; jt < 16; jt += 8) {
    v8f acc = {};
#pragma unroll
    for (int kk = 0; kk < 64; kk += 32)
      acc = wmma16(frag_ld(xln + jt * 16 * 64 + kk, 64, 1),
                   frag_ld(wsm + kk * 16, 1, 16), acc);
    if (n < 4) {
#pragma unroll
      for (int r = 0; r < 8; ++r) {
        int j = jt * 16 + r + hi * 8;
        p.bias[(((size_t)t * HH + n) * NN + i) * NN + j] = acc[r];
      }
    }
  }
}

// ---------------------------------------------------------------------------
// Triangle attention: phase B (logits, softmax, AV, gate, output proj)
// K/V/Q tiles staged into LDS by the Tensor Data Mover.
// ---------------------------------------------------------------------------
struct AttnCoreP {
  float* s; const float* mask;
  const _Float16* qf; const _Float16* kf; const _Float16* vf; const _Float16* gf;
  const float* bias; const float* wo; const float* bo;
  int tr;
};

// LDS partition byte offsets (no static __shared__ in this kernel, so dynamic
// LDS starts at offset 0 -> these are hardware LDS addresses for TDM).
#define L_KHH 0u                          // [256][32] f16 (K padded)
#define L_VHH (L_KHH + 256u * 32u * 2u)   // [256][16] f16
#define L_QHH (L_VHH + 256u * 16u * 2u)   // [64][32] f16
#define L_OG  (L_QHH + 64u * 32u * 2u)    // [64][32] f16
#define L_WOM (L_OG + 64u * 32u * 2u)     // [32][64] f16
#define L_LG  (L_WOM + 32u * 64u * 2u)    // [64][256] f32
#define L_MB  (L_LG + 64u * 256u * 4u)    // [256] f32
#define L_RED (L_MB + 256u * 4u)          // [256] f32
#define L_END (L_RED + 256u * 4u)

extern "C" __global__ __launch_bounds__(256) void k_attn_core(AttnCoreP p) {
  extern __shared__ char smem[];
  _Float16* khh = (_Float16*)(smem + L_KHH);
  _Float16* vhh = (_Float16*)(smem + L_VHH);
  _Float16* qhh = (_Float16*)(smem + L_QHH);
  _Float16* og  = (_Float16*)(smem + L_OG);
  _Float16* wom = (_Float16*)(smem + L_WOM);
  float* lg  = (float*)(smem + L_LG);
  float* mb  = (float*)(smem + L_MB);
  float* red = (float*)(smem + L_RED);

  int b = blockIdx.x;
  int js = b & 3;  b >>= 2;
  int h  = b & 3;  b >>= 2;
  int i  = b & 255; b >>= 8;
  int t  = b;
  int tid = threadIdx.x;
  int j0 = js * 64;
  int wid = tid >> 5, lane = tid & 31, n = lane & 15, hi = lane >> 4;

  // Zero the K-pad halves (TDM padding skips bytes, does not zero them).
  {
    int k = tid;
#pragma unroll
    for (int d = 0; d < 16; ++d) khh[k * 32 + 16 + d] = (_Float16)0.f;
    mb[k] = 1.0e9f *
        (p.mask[((size_t)t * NN + (p.tr ? k : i)) * NN + (p.tr ? i : k)] - 1.0f);
  }
  if (tid < 64) {
    int jj = tid;
#pragma unroll
    for (int d = 0; d < 16; ++d) {
      qhh[jj * 32 + 16 + d] = (_Float16)0.f;
      og[jj * 32 + 16 + d] = (_Float16)0.f;
    }
  }
  for (int idx = tid; idx < 32 * 64; idx += 256) {
    int d = idx >> 6, c = idx & 63;
    wom[idx] = (d < 16) ? (_Float16)p.wo[(h * 16 + d) * 64 + c] : (_Float16)0.f;
  }

  // TDM: DMA the three tiles (rows of 32B, 128B pitch) into LDS; K/Q tiles use
  // LDS padding to land in the [rows][32-half] layout the WMMA fragments use.
  if (wid == 0) {
    size_t row0 = (((size_t)t * NN + i) * NN) * CC + (size_t)h * 16;
    unsigned long long ka = (unsigned long long)(size_t)(p.kf + row0);
    unsigned long long va = (unsigned long long)(size_t)(p.vf + row0);
    unsigned long long qa = (unsigned long long)(size_t)(p.qf + row0 + (size_t)j0 * CC);
    tdm_load(tdm_g0(L_KHH, ka), tdm_g1(TDM_PAD_W0, 32u, 256u, 32u, 256u, 128ull));
    tdm_load(tdm_g0(L_VHH, va), tdm_g1(0u,         32u, 256u, 32u, 256u, 128ull));
    tdm_load(tdm_g0(L_QHH, qa), tdm_g1(TDM_PAD_W0, 32u,  64u, 32u,  64u, 128ull));
    __builtin_amdgcn_s_wait_tensorcnt(0);
  }
  __syncthreads();

  // logits = q k^T + mask_bias + tri_bias
  for (int tl = wid; tl < 64; tl += 8) {
    int jt = tl >> 4, kt = tl & 15;
    v8f acc = {};
    acc = wmma16(frag_ld(qhh + jt * 16 * 32, 32, 1),
                 frag_ld(khh + kt * 16 * 32, 32, 1), acc);  // B^T form
    int kc = kt * 16 + n;
#pragma unroll
    for (int r = 0; r < 8; ++r) {
      int jj = jt * 16 + r + hi * 8;
      lg[jj * 256 + kc] = acc[r] + mb[kc] +
          p.bias[(((size_t)t * HH + h) * NN + (j0 + jj)) * NN + kc];
    }
  }
  __syncthreads();

  // row softmax over k: 4 threads per row, LDS tree reduction
  {
    int row = tid >> 2, sub = tid & 3;
    float* r = lg + row * 256 + sub * 64;
    float mx = -3.0e38f;
    for (int k = 0; k < 64; ++k) mx = fmaxf(mx, r[k]);
    red[tid] = mx;
    __syncthreads();
    float m4 = fmaxf(fmaxf(red[row * 4 + 0], red[row * 4 + 1]),
                     fmaxf(red[row * 4 + 2], red[row * 4 + 3]));
    __syncthreads();
    float sum = 0.f;
    for (int k = 0; k < 64; ++k) { float e = __expf(r[k] - m4); r[k] = e; sum += e; }
    red[tid] = sum;
    __syncthreads();
    float s4 = (red[row * 4 + 0] + red[row * 4 + 1]) +
               (red[row * 4 + 2] + red[row * 4 + 3]);
    float inv = 1.0f / s4;
    for (int k = 0; k < 64; ++k) r[k] *= inv;
  }
  __syncthreads();

  // o = a @ v, then gate
  if (wid < 4) {
    int jt = wid;
    v8f acc = {};
#pragma unroll
    for (int kc = 0; kc < 8; ++kc)
      acc = wmma16(frag_ld_f32(lg + jt * 16 * 256 + kc * 32, 256, 1),
                   frag_ld(vhh + kc * 32 * 16, 1, 16), acc);
#pragma unroll
    for (int r = 0; r < 8; ++r) {
      int jj = jt * 16 + r + hi * 8;
      float gv = (float)p.gf[(((size_t)t * NN + i) * NN + (j0 + jj)) * CC + h * 16 + n];
      og[jj * 32 + n] = (_Float16)(acc[r] * gv);
    }
  }
  __syncthreads();

  // out = (o*g) @ wo_h   (+ bo once, on h==0), residual-add into s
  for (int tl = wid; tl < 16; tl += 8) {
    int jt = tl >> 2, ct = tl & 3;
    v8f acc = {};
    acc = wmma16(frag_ld(og + jt * 16 * 32, 32, 1),
                 frag_ld(wom + ct * 16, 1, 64), acc);
    int c = ct * 16 + n;
#pragma unroll
    for (int r = 0; r < 8; ++r) {
      int jg = j0 + jt * 16 + r + hi * 8;
      float v = acc[r] + ((h == 0) ? p.bo[c] : 0.f);
      atomicAdd(p.s + (((size_t)t * NN + (p.tr ? jg : i)) * NN +
                       (p.tr ? i : jg)) * CC + c, v);
    }
  }
}

// ---------------------------------------------------------------------------
// Triangle multiplication: phase A (LN + gated projections -> channel-major)
// ---------------------------------------------------------------------------
struct MulPrepP {
  const float* s; const float* mask;
  const float* ln_s; const float* ln_b;
  const float* wap; const float* bap; const float* wag; const float* bag;
  const float* wbp; const float* bbp; const float* wbg; const float* bbg;
  const float* wgp; const float* bgp;
  _Float16* acm; _Float16* bcm; _Float16* gws;
};

extern "C" __global__ __launch_bounds__(256) void k_mul_prep(MulPrepP p) {
  extern __shared__ char smem[];
  _Float16* zl = (_Float16*)smem;    // [256][64]
  _Float16* t0 = zl + NN * CC;       // [256][64]
  _Float16* t1 = t0 + NN * CC;       // [256][64]
  _Float16* wsm = t1 + NN * CC;      // [64][64]
  int t = blockIdx.x >> 8, i = blockIdx.x & 255;
  int tid = threadIdx.x;
  {
    int j = tid;
    const float* row = p.s + (((size_t)t * NN + i) * NN + j) * CC;
    layernorm_row(row, p.ln_s, p.ln_b, zl + (size_t)j * 64);
  }
  __syncthreads();

  // a = mask * sigmoid(zl@wag+bag) * (zl@wap+bap), channel-major [T][C][N][N]
  stage_w(p.wag, wsm, 64 * 64, tid); __syncthreads();
  projG(zl, wsm, t0, 1, p.bag); __syncthreads();
  stage_w(p.wap, wsm, 64 * 64, tid); __syncthreads();
  projG(zl, wsm, t1, 0, p.bap); __syncthreads();
  for (int idx = tid; idx < 256 * 64; idx += 256) {
    int j = idx >> 6, c = idx & 63;
    float mv = p.mask[((size_t)t * NN + i) * NN + j];
    p.acm[(((size_t)t * CC + c) * NN + i) * NN + j] =
        (_Float16)(mv * (float)t0[idx] * (float)t1[idx]);
  }
  __syncthreads();

  stage_w(p.wbg, wsm, 64 * 64, tid); __syncthreads();
  projG(zl, wsm, t0, 1, p.bbg); __syncthreads();
  stage_w(p.wbp, wsm, 64 * 64, tid); __syncthreads();
  projG(zl, wsm, t1, 0, p.bbp); __syncthreads();
  for (int idx = tid; idx < 256 * 64; idx += 256) {
    int j = idx >> 6, c = idx & 63;
    float mv = p.mask[((size_t)t * NN + i) * NN + j];
    p.bcm[(((size_t)t * CC + c) * NN + i) * NN + j] =
        (_Float16)(mv * (float)t0[idx] * (float)t1[idx]);
  }
  __syncthreads();

  // g = sigmoid(zl@wg+bg), row-major
  stage_w(p.wgp, wsm, 64 * 64, tid); __syncthreads();
  projG(zl, wsm, t0, 1, p.bgp); __syncthreads();
  size_t rowbase = (((size_t)t * NN + i) * NN) * CC;
  for (int idx = tid; idx < 256 * 64; idx += 256) p.gws[rowbase + idx] = t0[idx];
}

// ---------------------------------------------------------------------------
// Triangle multiplication: per-channel 256x256x256 f16 GEMM
// ---------------------------------------------------------------------------
struct MulMMP { const _Float16* acm; const _Float16* bcm; float* xbuf; int outgoing; };

extern "C" __global__ __launch_bounds__(256) void k_mul_mm(MulMMP p) {
  int b = blockIdx.x;
  int itg = b & 15; b >>= 4;
  int c   = b & 63; b >>= 6;
  int t   = b;
  int i0 = itg * 16;
  int wid = threadIdx.x >> 5, lane = threadIdx.x & 31, n = lane & 15, hi = lane >> 4;
  const _Float16* ab = p.acm + ((size_t)(t * CC + c)) * NN * NN;
  const _Float16* bb = p.bcm + ((size_t)(t * CC + c)) * NN * NN;
  for (int jt = wid; jt < 16; jt += 8) {
    if (jt + 8 < 16)   // prefetch next B tile-row (global_prefetch_b8)
      __builtin_prefetch(bb + (size_t)(jt + 8) * 16 * NN, 0, 1);
    v8f acc = {};
    if (p.outgoing) {        // x[i][j] = sum_k A[i][k] B[j][k]
#pragma unroll
      for (int kc = 0; kc < 8; ++kc)
        acc = wmma16(frag_ld(ab + (size_t)i0 * NN + kc * 32, NN, 1),
                     frag_ld(bb + (size_t)jt * 16 * NN + kc * 32, NN, 1), acc);
    } else {                 // x[i][j] = sum_k A[k][i] B[k][j]
#pragma unroll
      for (int kc = 0; kc < 8; ++kc)
        acc = wmma16(frag_ld(ab + (size_t)kc * 32 * NN + i0, 1, NN),
                     frag_ld(bb + (size_t)kc * 32 * NN + jt * 16, 1, NN), acc);
    }
#pragma unroll
    for (int r = 0; r < 8; ++r) {
      int ii = i0 + r + hi * 8;
      int jj = jt * 16 + n;
      p.xbuf[(((size_t)t * NN + ii) * NN + jj) * CC + c] = acc[r];
    }
  }
}

// ---------------------------------------------------------------------------
// Triangle multiplication: LN(out) + output proj * gate, residual add
// ---------------------------------------------------------------------------
struct MulOutP {
  float* s; const float* xbuf; const _Float16* gws;
  const float* ln_s; const float* ln_b; const float* wo; const float* bo;
};

extern "C" __global__ __launch_bounds__(256) void k_mul_out(MulOutP p) {
  extern __shared__ char smem[];
  _Float16* xl = (_Float16*)smem;   // [256][64]
  _Float16* wsm = xl + NN * CC;     // [64][64]
  int t = blockIdx.x >> 8, i = blockIdx.x & 255;
  int tid = threadIdx.x;
  {
    int j = tid;
    const float* row = p.xbuf + (((size_t)t * NN + i) * NN + j) * CC;
    layernorm_row(row, p.ln_s, p.ln_b, xl + (size_t)j * 64);
  }
  __syncthreads();
  stage_w(p.wo, wsm, 64 * 64, tid);
  __syncthreads();
  int wid = tid >> 5, lane = tid & 31, n = lane & 15, hi = lane >> 4;
  for (int tl = wid; tl < 64; tl += 8) {
    int jt = tl >> 2, nt = tl & 3;
    v8f acc = {};
#pragma unroll
    for (int kk = 0; kk < 64; kk += 32)
      acc = wmma16(frag_ld(xl + jt * 16 * 64 + kk, 64, 1),
                   frag_ld(wsm + kk * 64 + nt * 16, 1, 64), acc);
    int col = nt * 16 + n;
#pragma unroll
    for (int r = 0; r < 8; ++r) {
      int j = jt * 16 + r + hi * 8;
      size_t gi = (((size_t)t * NN + i) * NN + j) * CC + col;
      p.s[gi] += (acc[r] + p.bo[col]) * (float)p.gws[gi];
    }
  }
}

// ---------------------------------------------------------------------------
// Pair transition: LN -> 64x128 relu -> 128x64, masked residual add
// ---------------------------------------------------------------------------
struct TransP {
  float* s; const float* mask;
  const float* ln_s; const float* ln_b;
  const float* w1; const float* b1; const float* w2; const float* b2;
};

extern "C" __global__ __launch_bounds__(256) void k_trans(TransP p) {
  extern __shared__ char smem[];
  _Float16* zl = (_Float16*)smem;     // [256][64]
  _Float16* h1 = zl + NN * CC;        // [256][128]
  _Float16* wsm = h1 + NN * 128;      // [128][64] max
  int t = blockIdx.x >> 8, i = blockIdx.x & 255;
  int tid = threadIdx.x;
  {
    int j = tid;
    const float* row = p.s + (((size_t)t * NN + i) * NN + j) * CC;
    layernorm_row(row, p.ln_s, p.ln_b, zl + (size_t)j * 64);
  }
  __syncthreads();
  stage_w(p.w1, wsm, 64 * 128, tid);
  __syncthreads();
  int wid = tid >> 5, lane = tid & 31, n = lane & 15, hi = lane >> 4;
  for (int tl = wid; tl < 128; tl += 8) {
    int jt = tl >> 3, nt = tl & 7;
    v8f acc = {};
#pragma unroll
    for (int kk = 0; kk < 64; kk += 32)
      acc = wmma16(frag_ld(zl + jt * 16 * 64 + kk, 64, 1),
                   frag_ld(wsm + kk * 128 + nt * 16, 1, 128), acc);
    int col = nt * 16 + n;
#pragma unroll
    for (int r = 0; r < 8; ++r) {
      int j = jt * 16 + r + hi * 8;
      h1[(size_t)j * 128 + col] = (_Float16)fmaxf(acc[r] + p.b1[col], 0.f);
    }
  }
  __syncthreads();
  stage_w(p.w2, wsm, 128 * 64, tid);
  __syncthreads();
  for (int tl = wid; tl < 64; tl += 8) {
    int jt = tl >> 2, ct = tl & 3;
    v8f acc = {};
#pragma unroll
    for (int kk = 0; kk < 128; kk += 32)
      acc = wmma16(frag_ld(h1 + jt * 16 * 128 + kk, 128, 1),
                   frag_ld(wsm + kk * 64 + ct * 16, 1, 64), acc);
    int col = ct * 16 + n;
#pragma unroll
    for (int r = 0; r < 8; ++r) {
      int j = jt * 16 + r + hi * 8;
      float mv = p.mask[((size_t)t * NN + i) * NN + j];
      p.s[(((size_t)t * NN + i) * NN + j) * CC + col] +=
          (acc[r] + p.b2[col]) * mv;
    }
  }
}

// ---------------------------------------------------------------------------
// Host orchestration
// ---------------------------------------------------------------------------
// Input flattening (setup_inputs insertion order):
//  0 t, 1 mask,
//  att_start: 2 ln_s, 3 ln_b, 4 wq, 5 wk, 6 wv, 7 wb, 8 wg, 9 bg, 10 wo, 11 bo
//  att_end:   12..21 (same order)
//  mul_out:   22 ln_in_s, 23 ln_in_b, 24 wap, 25 bap, 26 wag, 27 bag,
//             28 wbp, 29 bbp, 30 wbg, 31 bbg, 32 wg, 33 bg,
//             34 ln_out_s, 35 ln_out_b, 36 wo, 37 bo
//  mul_in:    38..53 (same order)
//  trans:     54 ln_s, 55 ln_b, 56 w1, 57 b1, 58 w2, 59 b2
extern "C" void kernel_launch(void* const* d_in, const int* in_sizes, int n_in,
                              void* d_out, int out_size, void* d_ws, size_t ws_size,
                              hipStream_t stream) {
  (void)in_sizes; (void)n_in; (void)out_size; (void)ws_size;
  const float* t_in = (const float*)d_in[0];
  const float* mask = (const float*)d_in[1];
  float* s = (float*)d_out;

  const size_t ELEMS = (size_t)TT * NN * NN * CC;   // 8,388,608
  _Float16* qf = (_Float16*)d_ws;       // also acm for tri-mul
  _Float16* kf = qf + ELEMS;            // also bcm
  _Float16* vf = kf + ELEMS;
  _Float16* gf = vf + ELEMS;            // also tri-mul gate
  float* bias  = (float*)(gf + ELEMS);  // [T][H][N][N]
  float* xbuf  = bias + (size_t)TT * HH * NN * NN;

  hipMemcpyAsync(s, t_in, ELEMS * sizeof(float), hipMemcpyDeviceToDevice, stream);

  auto F = [&](int idx) { return (const float*)d_in[idx]; };

  const size_t sh_prep = (size_t)(NN * CC + CC * CC) * 2;                       // 40,960
  const size_t sh_core = L_END;                                                 // 104,448
  const size_t sh_mulp = (size_t)(3 * NN * CC + CC * CC) * 2;                   // 106,496
  const size_t sh_trans = (size_t)(NN * CC + NN * 128 + 128 * 64) * 2;          // 114,688

  // Triangle attention (starting, then ending node on transposed view)
  for (int stage = 0; stage < 2; ++stage) {
    int base = (stage == 0) ? 2 : 12;
    int tr = stage;
    AttnPrepP pp{ s, F(base+0), F(base+1), F(base+2), F(base+3), F(base+4),
                  F(base+5), F(base+6), F(base+7), qf, kf, vf, gf, bias, tr };
    k_attn_prep<<<dim3(TT * NN), dim3(256), sh_prep, stream>>>(pp);
    AttnCoreP cp{ s, mask, qf, kf, vf, gf, bias, F(base+8), F(base+9), tr };
    k_attn_core<<<dim3(TT * NN * HH * 4), dim3(256), sh_core, stream>>>(cp);
  }

  // Triangle multiplication (outgoing, then incoming)
  for (int stage = 0; stage < 2; ++stage) {
    int base = (stage == 0) ? 22 : 38;
    int outgoing = (stage == 0) ? 1 : 0;
    MulPrepP mp{ s, mask, F(base+0), F(base+1), F(base+2), F(base+3), F(base+4),
                 F(base+5), F(base+6), F(base+7), F(base+8), F(base+9),
                 F(base+10), F(base+11), qf, kf, gf };
    k_mul_prep<<<dim3(TT * NN), dim3(256), sh_mulp, stream>>>(mp);
    MulMMP mm{ qf, kf, xbuf, outgoing };
    k_mul_mm<<<dim3(TT * CC * 16), dim3(256), 0, stream>>>(mm);
    MulOutP mo{ s, xbuf, gf, F(base+12), F(base+13), F(base+14), F(base+15) };
    k_mul_out<<<dim3(TT * NN), dim3(256), sh_prep, stream>>>(mo);
  }

  // Pair transition
  TransP tp{ s, mask, F(54), F(55), F(56), F(57), F(58), F(59) };
  k_trans<<<dim3(TT * NN), dim3(256), sh_trans, stream>>>(tp);
}
